// FastMKA_53867479826759
// MI455X (gfx1250) — matrix operand order repaired and checked by reading, not compiled, not run on hardware
//
#include <hip/hip_runtime.h>
#include <math.h>

#define B_   2
#define T_   2048
#define C_   1024
#define H_   16
#define D_   64
#define MT_  (B_ * T_)          // 4096 rows
#define SCALE_ 0.125f           // D^-0.5
#define MP_CH 16                // mean-pool chunks over T

typedef __attribute__((ext_vector_type(16))) _Float16 v16h;
typedef __attribute__((ext_vector_type(8)))  _Float16 v8h;
typedef __attribute__((ext_vector_type(8)))  float    v8f;

union Frag { v16h v; v8h p[2]; _Float16 h[16]; };
union H8   { v8h v;  _Float16 h[8]; };

// ---------------------------------------------------------------- CDNA5 async global->LDS copy
#if __has_builtin(__builtin_amdgcn_global_load_async_to_lds_b128) && \
    __has_builtin(__builtin_amdgcn_s_wait_asynccnt)
#define HAVE_ASYNC_LDS 1
typedef int v4i __attribute__((vector_size(16)));              // matches builtin pointee
typedef __attribute__((address_space(1))) v4i g1_v4i;          // AS(1): global
typedef __attribute__((address_space(3))) v4i l3_v4i;          // AS(3): LDS
#endif

__device__ __forceinline__ void cp16(const _Float16* __restrict__ g, _Float16* l) {
#ifdef HAVE_ASYNC_LDS
  // flat LDS aperture keeps the LDS byte offset in addr[31:0] (ISA 10.2)
  __builtin_amdgcn_global_load_async_to_lds_b128(
      (g1_v4i*)(uintptr_t)g,
      (l3_v4i*)(uintptr_t)(unsigned)(uintptr_t)l, 0, 0);
#else
  *(v8h*)l = *(const v8h*)g;
#endif
}

__device__ __forceinline__ void async_wait0() {
#ifdef HAVE_ASYNC_LDS
  __builtin_amdgcn_s_wait_asynccnt(0);
#endif
}

// ---------------------------------------------------------------- CDNA5 Tensor Data Mover staging
#if __has_builtin(__builtin_amdgcn_tensor_load_to_lds) && \
    __has_builtin(__builtin_amdgcn_s_wait_tensorcnt)
#define HAVE_TDM 1
typedef unsigned int u32x4 __attribute__((vector_size(16)));
typedef int i32x8 __attribute__((vector_size(32)));
typedef int i32x4 __attribute__((vector_size(16)));

// Load rows x 64 f16 tile (global row stride = gstride elems) into LDS at lds_addr,
// rows padded to 72 halves (144B): pad_interval=4 (32 DW) pad_amount=3 (4 DW).
// D# per ISA 8.3/8.4: count=1, type=2(image), data_size=1(2B).
__device__ __forceinline__ void tdm_load_tile_f16(const _Float16* g, unsigned lds_addr,
                                                  unsigned rows, unsigned gstride) {
  unsigned long long ga = (unsigned long long)(uintptr_t)g;
  u32x4 g0 = { 1u,                                   // count=1, user mode
               lds_addr,                             // lds_addr [63:32]
               (unsigned)ga,                         // global_addr[31:0]
               ((unsigned)(ga >> 32) & 0x01FFFFFFu) | (2u << 30) };  // [56:32] | type=2
  i32x8 g1 = {
      (int)(0x10000u | (1u << 20) | (4u << 22) | (3u << 25)),  // data_size=1, pad en/int/amt
      (int)((gstride & 0xFFFFu) << 16),              // tensor_dim0[15:0] @ bits 63:48
      (int)((gstride >> 16) & 0xFFFFu),              // tensor_dim0[31:16]; tensor_dim1[15:0]=0
      (int)(16u | (64u << 16)),                      // tensor_dim1=1<<20 (no OOB); tile_dim0=64
      (int)(rows & 0xFFFFu),                         // tile_dim1=rows; tile_dim2=0
      (int)gstride,                                  // tensor_dim0_stride[31:0]
      0, 0 };
  i32x4 gz4 = { 0, 0, 0, 0 };                        // groups 2/3 unused (2D tile)
  i32x8 gz8 = { 0, 0, 0, 0, 0, 0, 0, 0 };            // trailing group (clang-23 6-arg form)
  __builtin_amdgcn_tensor_load_to_lds(g0, g1, gz4, gz4, gz8, 0);
}
#endif

// wait for this workgroup's staged tile (TDM issued by wave 0, else per-thread async)
__device__ __forceinline__ void stage_wait(int wave) {
#ifdef HAVE_TDM
  if (wave == 0) __builtin_amdgcn_s_wait_tensorcnt(0);
#else
  async_wait0();
#endif
  __syncthreads();
}

// ---------------------------------------------------------------- DPP16 half-wave (16-lane) reductions
#if __has_builtin(__builtin_amdgcn_update_dpp)
#define HAVE_DPP 1
#endif

template <int CTRL>
__device__ __forceinline__ float dppf(float x) {
#ifdef HAVE_DPP
  int xi = __float_as_int(x);
  xi = __builtin_amdgcn_update_dpp(xi, xi, CTRL, 0xf, 0xf, true);
  return __int_as_float(xi);
#else
  return x;
#endif
}

__device__ __forceinline__ float rowmax16(float x) {
#ifdef HAVE_DPP
  x = fmaxf(x, dppf<0xB1>(x));    // quad_perm xor1
  x = fmaxf(x, dppf<0x4E>(x));    // quad_perm xor2
  x = fmaxf(x, dppf<0x141>(x));   // row_half_mirror (xor7)
  x = fmaxf(x, dppf<0x140>(x));   // row_mirror (xor15)
#else
#pragma unroll
  for (int off = 1; off < 16; off <<= 1) x = fmaxf(x, __shfl_xor(x, off, 32));
#endif
  return x;
}

__device__ __forceinline__ float rowsum16(float x) {
#ifdef HAVE_DPP
  x += dppf<0xB1>(x);
  x += dppf<0x4E>(x);
  x += dppf<0x141>(x);
  x += dppf<0x140>(x);
#else
#pragma unroll
  for (int off = 1; off < 16; off <<= 1) x += __shfl_xor(x, off, 32);
#endif
  return x;
}

// ---------------------------------------------------------------- conversions (8-wide)
__global__ __launch_bounds__(256) void cvt_f32_f16(const float* __restrict__ in,
                                                   _Float16* __restrict__ out, int n8) {
  int i = blockIdx.x * 256 + threadIdx.x;
  if (i < n8) {
    const float4* p = (const float4*)in + (size_t)i * 2;
    float4 a = p[0], bq = p[1];
    H8 o;
    o.h[0] = (_Float16)a.x;  o.h[1] = (_Float16)a.y;
    o.h[2] = (_Float16)a.z;  o.h[3] = (_Float16)a.w;
    o.h[4] = (_Float16)bq.x; o.h[5] = (_Float16)bq.y;
    o.h[6] = (_Float16)bq.z; o.h[7] = (_Float16)bq.w;
    *(v8h*)(out + (size_t)i * 8) = o.v;
  }
}

// ---------------------------------------------------------------- mean pool over T (two stages)
__global__ __launch_bounds__(256) void mean_pool1(const float* __restrict__ x,
                                                  float* __restrict__ part) {
  int i = blockIdx.x * 256 + threadIdx.x;            // B_*MP_CH*C_ threads
  int c = i % C_, rest = i / C_;
  int b = rest / MP_CH, ch = rest % MP_CH;
  const int TC = T_ / MP_CH;
  float s = 0.f;
  for (int t = 0; t < TC; ++t) s += x[((size_t)b * T_ + ch * TC + t) * C_ + c];
  part[i] = s;
}

__global__ __launch_bounds__(256) void mean_pool2(const float* __restrict__ part,
                                                  float* __restrict__ meanp) {
  int i = blockIdx.x * 256 + threadIdx.x;            // B_*C_ threads
  int b = i / C_, c = i % C_;
  float s = 0.f;
#pragma unroll
  for (int ch = 0; ch < MP_CH; ++ch) s += part[((size_t)b * MP_CH + ch) * C_ + c];
  meanp[i] = s * (1.0f / T_);
}

// ---------------------------------------------------------------- WMMA GEMM: Y = act(X @ W^T + bias)
// X: [M,K] f16 row-major, W: [N,K] f16 row-major.
// 256 threads (8 waves): 128x128 tile, KC=64, double-buffered TDM/async staging.
template <int ACT, bool OUT32>
__global__ __launch_bounds__(256) void gemm_xwT(
    const _Float16* __restrict__ X, const _Float16* __restrict__ W,
    const float* __restrict__ bias,
    _Float16* __restrict__ Yh, float* __restrict__ Yf,
    int M, int N, int K) {
  __shared__ __align__(16) _Float16 sX[2][128][72];   // 64 + pad, 144B rows
  __shared__ __align__(16) _Float16 sW[2][128][72];

  const int tid  = threadIdx.x;
  const int lane = tid & 31, wave = tid >> 5;
  const int wm = wave & 3, wn = wave >> 2;
  const int m0 = blockIdx.y * 128, n0 = blockIdx.x * 128;
  const int lr = tid >> 3, lc = (tid & 7) * 8;        // fallback: 32 rows/pass, 8-half chunks

  auto loadStage = [&](int bufb, int kc) {
#ifdef HAVE_TDM
    if (wave == 0) {
      tdm_load_tile_f16(&X[(size_t)m0 * K + kc], (unsigned)(uintptr_t)&sX[bufb][0][0], 128, K);
      tdm_load_tile_f16(&W[(size_t)n0 * K + kc], (unsigned)(uintptr_t)&sW[bufb][0][0], 128, K);
    }
#else
#pragma unroll
    for (int p = 0; p < 4; ++p) {
      int row = lr + 32 * p;
      cp16(&X[(size_t)(m0 + row) * K + kc + lc], &sX[bufb][row][lc]);
      cp16(&W[(size_t)(n0 + row) * K + kc + lc], &sW[bufb][row][lc]);
      if (kc + 64 < K) {
        __builtin_prefetch(&X[(size_t)(m0 + row) * K + kc + 64 + lc], 0, 1);
        __builtin_prefetch(&W[(size_t)(n0 + row) * K + kc + 64 + lc], 0, 1);
      }
    }
#endif
  };

  v8f zero = {};
  v8f acc[2][4];
#pragma unroll
  for (int i = 0; i < 2; ++i)
#pragma unroll
    for (int j = 0; j < 4; ++j) acc[i][j] = zero;

  loadStage(0, 0);
  int buf = 0;
  for (int kc = 0; kc < K; kc += 64) {
    stage_wait(wave);
    if (kc + 64 < K) loadStage(buf ^ 1, kc + 64);

#pragma unroll
    for (int ks = 0; ks < 2; ++ks) {
      const int k = ks * 32;
      Frag a[2], b[4];
#pragma unroll
      for (int i = 0; i < 2; ++i) {                // A 16x32: lane m=L%16, K per ISA layout
        int mr = wm * 32 + i * 16 + (lane & 15);
        int ko = k + (lane >> 4) * 8;
        a[i].p[0] = *(const v8h*)&sX[buf][mr][ko];
        a[i].p[1] = *(const v8h*)&sX[buf][mr][ko + 16];
      }
#pragma unroll
      for (int j = 0; j < 4; ++j) {                // B 32x16: lane n=L%16, K per ISA layout
        int nr = wn * 64 + j * 16 + (lane & 15);
        int kb = k + (lane >> 4) * 8;
        b[j].p[0] = *(const v8h*)&sW[buf][nr][kb];
        b[j].p[1] = *(const v8h*)&sW[buf][nr][kb + 16];
      }
#pragma unroll
      for (int i = 0; i < 2; ++i)
#pragma unroll
        for (int j = 0; j < 4; ++j)
          acc[i][j] = __builtin_amdgcn_wmma_f32_16x16x32_f16(
              false, a[i].v, false, b[j].v, (short)0, acc[i][j], false, false);
    }
    buf ^= 1;
  }

  // epilogue: C layout -> lane L vgpr r holds (M = r + 8*(L/16), N = L%16)
#pragma unroll
  for (int i = 0; i < 2; ++i)
#pragma unroll
    for (int j = 0; j < 4; ++j)
#pragma unroll
      for (int r = 0; r < 8; ++r) {
        int row = m0 + wm * 32 + i * 16 + r + 8 * (lane >> 4);
        int col = n0 + wn * 64 + j * 16 + (lane & 15);
        float v = acc[i][j][r] + bias[col];
        if (ACT == 1) v = tanhf(v);
        if (OUT32) Yf[(size_t)row * N + col] = v;
        else       Yh[(size_t)row * N + col] = (_Float16)v;
      }
}

// ---------------------------------------------------------------- routing softmax + routed memory
__global__ __launch_bounds__(256) void routing_kernel(
    const _Float16* __restrict__ r1h, const float* __restrict__ x,
    const float* __restrict__ Wr2, const float* __restrict__ br2,
    const float* __restrict__ meanp, _Float16* __restrict__ routed) {
  const int lane = threadIdx.x & 31;
  const int row  = blockIdx.x * 8 + (threadIdx.x >> 5);   // one wave per token
  float a0 = 0.f, a1 = 0.f, a2 = 0.f;
  for (int c = lane; c < C_; c += 32) {
    float rv = (float)r1h[(size_t)row * C_ + c];
    a0 += rv * Wr2[c];
    a1 += rv * Wr2[C_ + c];
    a2 += rv * Wr2[2 * C_ + c];
  }
#pragma unroll
  for (int off = 16; off >= 1; off >>= 1) {
    a0 += __shfl_xor(a0, off, 32);
    a1 += __shfl_xor(a1, off, 32);
    a2 += __shfl_xor(a2, off, 32);
  }
  a0 += br2[0]; a1 += br2[1]; a2 += br2[2];
  float mx = fmaxf(a0, fmaxf(a1, a2));
  float e0 = __expf(a0 - mx), e1 = __expf(a1 - mx), e2 = __expf(a2 - mx);
  float inv = 1.0f / (e0 + e1 + e2);
  float w0 = e0 * inv, w1 = e1 * inv;
  const int b = row / T_;
  for (int c = lane; c < C_; c += 32)
    routed[(size_t)row * C_ + c] =
        (_Float16)(w0 * x[(size_t)row * C_ + c] + w1 * meanp[b * C_ + c]);
}

// ---------------------------------------------------------------- flash attention (per b,h; 128 q-rows / block)
// 64-key tiles, double-buffered: K via TDM/async, V via register transpose.
__global__ __launch_bounds__(256) void flash_attn(
    const _Float16* __restrict__ qh, const _Float16* __restrict__ kh,
    const _Float16* __restrict__ vh, _Float16* __restrict__ oh) {
  __shared__ __align__(16) _Float16 sK[2][64][72];     // keys x d (row-major, padded)
  __shared__ __align__(16) _Float16 sVt[2][64][72];    // d x keys (transposed, padded)
  __shared__ __align__(16) _Float16 sP[8][16][72];     // per-wave P tile (16q x 64k)

  const int tid = threadIdx.x, lane = tid & 31, wave = tid >> 5;
  const int b = blockIdx.z, h = blockIdx.y;
  const size_t base = (size_t)b * T_ * C_ + (size_t)h * D_;
  const int q0 = blockIdx.x * 128 + wave * 16;
  const int m = lane & 15, g = lane >> 4;
  const int krow = tid >> 3, kch = (tid & 7) * 8;

  auto stageKV = [&](int bufb, int kt) {
#ifdef HAVE_TDM
    if (wave == 0)
      tdm_load_tile_f16(&kh[base + (size_t)kt * C_],
                        (unsigned)(uintptr_t)&sK[bufb][0][0], 64, C_);
#endif
#pragma unroll
    for (int p = 0; p < 2; ++p) {
      int row = krow + 32 * p;
#ifndef HAVE_TDM
      cp16(&kh[base + (size_t)(kt + row) * C_ + kch], &sK[bufb][row][kch]);
#endif
      H8 vt; vt.v = *(const v8h*)&vh[base + (size_t)(kt + row) * C_ + kch];
#pragma unroll
      for (int t = 0; t < 8; ++t) sVt[bufb][kch + t][row] = vt.h[t];
    }
  };

  // Q fragments (A layout), held for the whole key loop: d-chunks [0,32),[32,64)
  Frag qa[2];
#pragma unroll
  for (int i = 0; i < 2; ++i) {
    const _Float16* qp = &qh[base + (size_t)(q0 + m) * C_ + i * 32 + g * 8];
    qa[i].p[0] = *(const v8h*)qp;
    qa[i].p[1] = *(const v8h*)(qp + 16);
  }

  v8f zero = {};
  v8f oacc[4];
#pragma unroll
  for (int a = 0; a < 4; ++a) oacc[a] = zero;
  float rm[8], rl[8];
#pragma unroll
  for (int r = 0; r < 8; ++r) { rm[r] = -3.0e38f; rl[r] = 0.f; }

  stageKV(0, 0);
  int buf = 0;
  for (int kt = 0; kt < T_; kt += 64) {
    stage_wait(wave);
    if (kt + 64 < T_) stageKV(buf ^ 1, kt + 64);

    // S = Q @ K^T : four 16-key column tiles, K-dim 64 in two WMMA steps each
    v8f sacc[4] = {zero, zero, zero, zero};
#pragma unroll
    for (int n = 0; n < 4; ++n)
#pragma unroll
      for (int dc = 0; dc < 2; ++dc) {
        Frag bk_;
        bk_.p[0] = *(const v8h*)&sK[buf][n * 16 + m][dc * 32 + g * 8];
        bk_.p[1] = *(const v8h*)&sK[buf][n * 16 + m][dc * 32 + g * 8 + 16];
        sacc[n] = __builtin_amdgcn_wmma_f32_16x16x32_f16(
            false, qa[dc].v, false, bk_.v, (short)0, sacc[n], false, false);
      }

    // online softmax per query row (rows live in 16-lane half-wave groups)
#pragma unroll
    for (int r = 0; r < 8; ++r) {
      float xs[4];
      float mx = -3.0e38f;
#pragma unroll
      for (int n = 0; n < 4; ++n) { xs[n] = sacc[n][r] * SCALE_; mx = fmaxf(mx, xs[n]); }
      mx = rowmax16(mx);
      float mnew = fmaxf(rm[r], mx);
      float corr = __expf(rm[r] - mnew);
      float ss = 0.f;
#pragma unroll
      for (int n = 0; n < 4; ++n) {
        float p = __expf(xs[n] - mnew);
        sP[wave][r + 8 * g][n * 16 + m] = (_Float16)p;   // C layout -> LDS
        ss += p;
      }
      ss = rowsum16(ss);
      rl[r] = rl[r] * corr + ss;
      rm[r] = mnew;
#pragma unroll
      for (int a = 0; a < 4; ++a) oacc[a][r] *= corr;
    }
    __syncthreads();

    // O += P @ V (P re-read in A layout; V^T gives contiguous B fragments)
    Frag pa[2];
#pragma unroll
    for (int kc2 = 0; kc2 < 2; ++kc2) {
      pa[kc2].p[0] = *(const v8h*)&sP[wave][m][kc2 * 32 + g * 8];
      pa[kc2].p[1] = *(const v8h*)&sP[wave][m][kc2 * 32 + g * 8 + 16];
    }
#pragma unroll
    for (int a = 0; a < 4; ++a)
#pragma unroll
      for (int kc2 = 0; kc2 < 2; ++kc2) {
        Frag bv_;
        bv_.p[0] = *(const v8h*)&sVt[buf][a * 16 + m][kc2 * 32 + g * 8];
        bv_.p[1] = *(const v8h*)&sVt[buf][a * 16 + m][kc2 * 32 + g * 8 + 16];
        oacc[a] = __builtin_amdgcn_wmma_f32_16x16x32_f16(
            false, pa[kc2].v, false, bv_.v, (short)0, oacc[a], false, false);
      }
    buf ^= 1;
  }

  // normalize and store [B,T,H*D] f16
#pragma unroll
  for (int r = 0; r < 8; ++r) {
    float inv = 1.0f / rl[r];
    int row = q0 + r + 8 * g;
#pragma unroll
    for (int a = 0; a < 4; ++a) {
      int col = h * D_ + a * 16 + m;
      oh[(size_t)(b * T_ + row) * C_ + col] = (_Float16)(oacc[a][r] * inv);
    }
  }
}

// ---------------------------------------------------------------- host launcher
extern "C" void kernel_launch(void* const* d_in, const int* in_sizes, int n_in,
                              void* d_out, int out_size, void* d_ws, size_t ws_size,
                              hipStream_t stream) {
  (void)in_sizes; (void)n_in; (void)out_size; (void)ws_size;
  const float* x   = (const float*)d_in[0];
  const float* Wq  = (const float*)d_in[1];
  const float* bq  = (const float*)d_in[2];
  const float* Wk  = (const float*)d_in[3];
  const float* bk  = (const float*)d_in[4];
  const float* Wv  = (const float*)d_in[5];
  const float* bv  = (const float*)d_in[6];
  const float* Wo  = (const float*)d_in[7];
  const float* bo  = (const float*)d_in[8];
  const float* Wr1 = (const float*)d_in[9];
  const float* br1 = (const float*)d_in[10];
  const float* Wr2 = (const float*)d_in[11];
  const float* br2 = (const float*)d_in[12];

  char* ws = (char*)d_ws;
  size_t off = 0;
  auto take = [&](size_t bytes) -> void* {
    void* p = ws + off;
    off += (bytes + 255) & ~(size_t)255;
    return p;
  };
  _Float16* xh      = (_Float16*)take((size_t)MT_ * C_ * 2);
  _Float16* Wqh     = (_Float16*)take((size_t)C_ * C_ * 2);
  _Float16* Wkh     = (_Float16*)take((size_t)C_ * C_ * 2);
  _Float16* Wvh     = (_Float16*)take((size_t)C_ * C_ * 2);
  _Float16* Woh     = (_Float16*)take((size_t)C_ * C_ * 2);
  _Float16* Wr1h    = (_Float16*)take((size_t)C_ * C_ * 2);
  _Float16* r1h     = (_Float16*)take((size_t)MT_ * C_ * 2);  // reused as attn-out later
  _Float16* routedh = (_Float16*)take((size_t)MT_ * C_ * 2);
  _Float16* qh      = (_Float16*)take((size_t)MT_ * C_ * 2);
  _Float16* kh      = (_Float16*)take((size_t)MT_ * C_ * 2);
  _Float16* vh      = (_Float16*)take((size_t)MT_ * C_ * 2);
  float*    meanp   = (float*)take((size_t)B_ * C_ * 4);
  float*    mpart   = (float*)take((size_t)B_ * MP_CH * C_ * 4);

  int n8 = (MT_ * C_) / 8;
  cvt_f32_f16<<<(n8 + 255) / 256, 256, 0, stream>>>(x, xh, n8);
  n8 = (C_ * C_) / 8;
  cvt_f32_f16<<<(n8 + 255) / 256, 256, 0, stream>>>(Wq,  Wqh,  n8);
  cvt_f32_f16<<<(n8 + 255) / 256, 256, 0, stream>>>(Wk,  Wkh,  n8);
  cvt_f32_f16<<<(n8 + 255) / 256, 256, 0, stream>>>(Wv,  Wvh,  n8);
  cvt_f32_f16<<<(n8 + 255) / 256, 256, 0, stream>>>(Wo,  Woh,  n8);
  cvt_f32_f16<<<(n8 + 255) / 256, 256, 0, stream>>>(Wr1, Wr1h, n8);

  mean_pool1<<<(B_ * MP_CH * C_) / 256, 256, 0, stream>>>(x, mpart);
  mean_pool2<<<(B_ * C_) / 256, 256, 0, stream>>>(mpart, meanp);

  dim3 gg(C_ / 128, MT_ / 128);   // (8, 32)
  // r1 = tanh(x @ Wr1^T + br1)
  gemm_xwT<1, false><<<gg, 256, 0, stream>>>(xh, Wr1h, br1, r1h, nullptr, MT_, C_, C_);
  // routing softmax -> routed memory (f16)
  routing_kernel<<<MT_ / 8, 256, 0, stream>>>(r1h, x, Wr2, br2, meanp, routedh);
  // q/k/v projections
  gemm_xwT<0, false><<<gg, 256, 0, stream>>>(xh,      Wqh, bq, qh, nullptr, MT_, C_, C_);
  gemm_xwT<0, false><<<gg, 256, 0, stream>>>(routedh, Wkh, bk, kh, nullptr, MT_, C_, C_);
  gemm_xwT<0, false><<<gg, 256, 0, stream>>>(routedh, Wvh, bv, vh, nullptr, MT_, C_, C_);
  // flash attention -> r1h (reused as attention output, f16)
  flash_attn<<<dim3(T_ / 128, H_, B_), 256, 0, stream>>>(qh, kh, vh, r1h);
  // final projection, f32 output
  gemm_xwT<0, true><<<gg, 256, 0, stream>>>(r1h, Woh, bo, nullptr, (float*)d_out, MT_, C_, C_);
}
